// LarvalTransportModule_29300266893566
// MI455X (gfx1250) — matrix operand order repaired and checked by reading, not compiled, not run on hardware
//
#include <hip/hip_runtime.h>
#include <hip/hip_bf16.h>
#include <math.h>

// ---------------------------------------------------------------------------
// FastKAN larval-transport graph layer for MI455X (gfx1250, wave32).
// The 32x32 edge-feature linear layer is done with v_wmma_f32_16x16x32_f16
// (16-edge tiles per wave); everything else is TRANS(v_exp_f32)-bound VALU.
// ---------------------------------------------------------------------------

typedef __attribute__((ext_vector_type(16))) _Float16 v16h;
typedef __attribute__((ext_vector_type(8)))  float    v8f;

#define NB 8
#define HDIM 32
__device__ __forceinline__ float rbf_grid(int b) {
    // jnp.linspace(-2,2,8): step 4/7
    return -2.0f + (float)b * (4.0f / 7.0f);
}
#define INV_DENOM 1.75f   // 1/(4/7)

__device__ __forceinline__ void atomicMaxF(float* addr, float val) {
    // float-max via monotone integer reinterpretation (array init'd to -FLT_MAX)
    if (val >= 0.0f) atomicMax((int*)addr, __float_as_int(val));
    else             atomicMin((unsigned int*)addr, __float_as_uint(val));
}

// ---------------------------------------------------------------------------
__global__ void lt_init_kernel(float* __restrict__ pressure,
                               float* __restrict__ att_sum,
                               float* __restrict__ att_max, int N) {
    int i = blockIdx.x * blockDim.x + threadIdx.x;
    if (i < N * HDIM) pressure[i] = 0.0f;
    if (i < N) { att_sum[i] = 0.0f; att_max[i] = -3.402823466e38f; }
}

// ---------------------------------------------------------------------------
// Kernel 1: per-wave tile of 16 edges.
//   A (16x32 f16)  : RBF expansion of normalized edge attrs (lane = edge row)
//   B (32x16 f16)  : W_edge^T columns (lane = out feature, k by lane-half)
//   C/D (16x16 f32): edge_features, bias preloaded in C
// Then attention partials from the D layout + h_src gather, half-wave
// shfl_xor reduction, larvae + softplus, atomic segment-max.
// ---------------------------------------------------------------------------
__global__ void __launch_bounds__(256)
lt_edge_kernel(const float* __restrict__ h,
               const int*   __restrict__ src,
               const int*   __restrict__ dst,
               const float* __restrict__ edge_attr,
               const float* __restrict__ afc,
               const float* __restrict__ W_edge,
               const float* __restrict__ b_edge,
               const float* __restrict__ W_att,
               const float* __restrict__ b_att,
               const float* __restrict__ W_lar,
               const float* __restrict__ b_lar,
               float* __restrict__ att_raw,
               float* __restrict__ larvae,
               float* __restrict__ att_max,
               int E) {
    const int tid  = threadIdx.x;
    const int wave = tid >> 5;
    const int lane = tid & 31;
    const int n    = lane & 15;     // column / edge-row selector
    const int hi   = lane >> 4;     // half-wave id
    const int base = (blockIdx.x * 8 + wave) * 16;
    if (base >= E) return;          // wave-uniform exit: EXEC stays all-ones

    // ---- A fragment: RBF(edge_attr) for edge row m = n --------------------
    int eA = base + n; if (eA >= E) eA = E - 1;
    float xf[4];
    xf[0] = edge_attr[eA * 4 + 0] * (1.0f / 50.0f);
    xf[1] = edge_attr[eA * 4 + 1];
    xf[2] = edge_attr[eA * 4 + 2];
    xf[3] = (edge_attr[eA * 4 + 3] - 10.0f) * 0.1f;
    v16h a;
#pragma unroll
    for (int j = 0; j < 16; ++j) {
        // contiguous k order (matches B): K = hi*16 + j
        int k = hi * 16 + j;
        float t = (xf[k >> 3] - rbf_grid(k & 7)) * INV_DENOM;
        a[j] = (_Float16)__expf(-t * t);
    }

    // ---- B fragments: B[k][n] = W_edge[n][k]; lane-half selects k block ----
    v16h b0, b1;
#pragma unroll
    for (int j = 0; j < 16; ++j) {
        int k = hi * 16 + j;
        b0[j] = (_Float16)W_edge[(n)      * HDIM + k];
        b1[j] = (_Float16)W_edge[(n + 16) * HDIM + k];
    }

    // ---- C = bias broadcast, two WMMAs for the 16x32 output tile ----------
    v8f c0, c1;
    float bias0 = b_edge[n], bias1 = b_edge[n + 16];
#pragma unroll
    for (int r = 0; r < 8; ++r) { c0[r] = bias0; c1[r] = bias1; }
    v8f d0 = __builtin_amdgcn_wmma_f32_16x16x32_f16(false, a, false, b0,
                                                    (short)0, c0, false, false);
    v8f d1 = __builtin_amdgcn_wmma_f32_16x16x32_f16(false, a, false, b1,
                                                    (short)0, c1, false, false);

    // ---- attention partials: this lane owns features n and n+16 of both
    //      edge_features and h_src, for edges m = hi*8 + r ------------------
    float wa0[NB], wa1[NB], wa2[NB], wa3[NB];
#pragma unroll
    for (int bb = 0; bb < NB; ++bb) {
        wa0[bb] = W_att[(n)      * NB + bb];
        wa1[bb] = W_att[(n + 16) * NB + bb];
        wa2[bb] = W_att[(32 + n) * NB + bb];
        wa3[bb] = W_att[(48 + n) * NB + bb];
    }
    float p[8];
    int   sidx[8];
#pragma unroll
    for (int r = 0; r < 8; ++r) {
        int e = base + hi * 8 + r; if (e >= E) e = E - 1;
        int s = src[e];
        sidx[r] = s;
        float v0  = d0[r];                 // edge_features[m][n]
        float v1  = d1[r];                 // edge_features[m][n+16]
        float hv0 = h[s * HDIM + n];       // h_src[m][n]
        float hv1 = h[s * HDIM + n + 16];  // h_src[m][n+16]
        float acc = 0.0f;
#pragma unroll
        for (int bb = 0; bb < NB; ++bb) {
            float g = rbf_grid(bb);
            float t0 = (v0  - g) * INV_DENOM; acc += __expf(-t0 * t0) * wa0[bb];
            float t1 = (v1  - g) * INV_DENOM; acc += __expf(-t1 * t1) * wa1[bb];
            float t2 = (hv0 - g) * INV_DENOM; acc += __expf(-t2 * t2) * wa2[bb];
            float t3 = (hv1 - g) * INV_DENOM; acc += __expf(-t3 * t3) * wa3[bb];
        }
        p[r] = acc;
    }
    // reduce over the 16 columns inside each half-wave (wave32 shuffles)
#pragma unroll
    for (int off = 1; off < 16; off <<= 1) {
#pragma unroll
        for (int r = 0; r < 8; ++r) p[r] += __shfl_xor(p[r], off, 32);
    }

    // ---- writers: lanes with n<8 own one edge each (e = base + hi*8 + n) --
    if (n < 8) {
        int e = base + hi * 8 + n;
        if (e < E) {
            float att = p[n] + b_att[0];
            att_raw[e] = att;
            // larvae: 8-RBF of adult_female_counts[src] -> softplus
            float xl = afc[sidx[n]];
            float acc = 0.0f;
#pragma unroll
            for (int bb = 0; bb < NB; ++bb) {
                float t = (xl - rbf_grid(bb)) * INV_DENOM;
                acc += __expf(-t * t) * W_lar[bb];
            }
            acc += b_lar[0];
            larvae[e] = (acc > 20.0f) ? acc : log1pf(__expf(acc));
            atomicMaxF(&att_max[dst[e]], att);
        }
    }
}

// ---------------------------------------------------------------------------
// Kernel 2: att_exp = exp(att_raw - att_max[dst]); segment-sum via atomics.
// ---------------------------------------------------------------------------
__global__ void lt_exp_kernel(const int* __restrict__ dst,
                              const float* __restrict__ att_max,
                              float* __restrict__ att_raw_exp,   // in: raw, out: exp
                              float* __restrict__ att_sum, int E) {
    int e = blockIdx.x * blockDim.x + threadIdx.x;
    if (e >= E) return;
    int d = dst[e];
    float ex = __expf(att_raw_exp[e] - att_max[d]);
    att_raw_exp[e] = ex;
    atomicAdd(&att_sum[d], ex);
}

// ---------------------------------------------------------------------------
// Kernel 3: wave per edge, lane per feature; coalesced float atomics into
// the L2-resident (N,32) pressure array.
// ---------------------------------------------------------------------------
__global__ void lt_scatter_kernel(const float* __restrict__ h,
                                  const int* __restrict__ src,
                                  const int* __restrict__ dst,
                                  const float* __restrict__ att_exp,
                                  const float* __restrict__ att_sum,
                                  const float* __restrict__ larvae,
                                  float* __restrict__ pressure, int E) {
    int g = blockIdx.x * blockDim.x + threadIdx.x;
    int e = g >> 5, lane = g & 31;
    if (e >= E) return;
    int d = dst[e], s = src[e];
    float w    = att_exp[e] / (att_sum[d] + 1e-8f);
    float coef = w * larvae[e];
    atomicAdd(&pressure[d * HDIM + lane], coef * h[s * HDIM + lane]);
}

// ---------------------------------------------------------------------------
extern "C" void kernel_launch(void* const* d_in, const int* in_sizes, int n_in,
                              void* d_out, int out_size, void* d_ws, size_t ws_size,
                              hipStream_t stream) {
    const float* h          = (const float*)d_in[0];   // (N,32)
    const int*   edge_index = (const int*)  d_in[1];   // (2,E)
    const float* edge_attr  = (const float*)d_in[2];   // (E,4)
    const float* afc        = (const float*)d_in[3];   // (N,)
    const float* W_edge     = (const float*)d_in[4];   // (32,32)
    const float* b_edge     = (const float*)d_in[5];   // (32,)
    const float* W_att      = (const float*)d_in[6];   // (1,512)
    const float* b_att      = (const float*)d_in[7];   // (1,)
    const float* W_lar      = (const float*)d_in[8];   // (1,8)
    const float* b_lar      = (const float*)d_in[9];   // (1,)
    float* pressure = (float*)d_out;                   // (N,32)

    const int N = in_sizes[0] / HDIM;
    const int E = in_sizes[2] / 4;
    const int* src = edge_index;
    const int* dst = edge_index + E;

    // workspace layout: att_raw/exp (E) | larvae (E) | att_sum (N) | att_max (N)
    float* att_raw = (float*)d_ws;
    float* larvae  = att_raw + E;
    float* att_sum = larvae + E;
    float* att_max = att_sum + N;

    const int BLK = 256;
    int initBlocks = (N * HDIM + BLK - 1) / BLK;
    lt_init_kernel<<<initBlocks, BLK, 0, stream>>>(pressure, att_sum, att_max, N);

    int edgeBlocks = (E + 128 - 1) / 128;   // 8 waves * 16 edges per block
    lt_edge_kernel<<<edgeBlocks, BLK, 0, stream>>>(
        h, src, dst, edge_attr, afc, W_edge, b_edge, W_att, b_att, W_lar, b_lar,
        att_raw, larvae, att_max, E);

    int expBlocks = (E + BLK - 1) / BLK;
    lt_exp_kernel<<<expBlocks, BLK, 0, stream>>>(dst, att_max, att_raw, att_sum, E);

    long long threads3 = (long long)E * 32;
    int scatBlocks = (int)((threads3 + BLK - 1) / BLK);
    lt_scatter_kernel<<<scatBlocks, BLK, 0, stream>>>(
        h, src, dst, att_raw, att_sum, larvae, pressure, E);
}
